// AudioDecoder_5111011082673
// MI455X (gfx1250) — matrix operand order repaired and checked by reading, not compiled, not run on hardware
//
#include <hip/hip_runtime.h>
#include <hip/hip_bf16.h>

// ---------------- problem constants ----------------
#define LATENT 128
#define FEAT   80
#define HID    256
#define G4     (4 * HID)   // 1024
#define MAXLEN 100
#define BATCH  4096
#define KIN    96          // input K padded 80 -> 96 (3 x 32)
#define K0TOT  (KIN + HID) // 352: layer0 LDS row extent
#define K1TOT  (2 * HID)   // 512: layer1 LDS row extent

typedef __bf16 bf16_t;
typedef __attribute__((ext_vector_type(16))) __bf16 v16bf;
typedef __attribute__((ext_vector_type(8)))  __bf16 v8bf;
typedef __attribute__((ext_vector_type(8)))  float  v8f;

// ---------------- workspace layout (bytes) ----------------
#define OFF_WIH0  0u           // 1024*96  bf16 = 196608
#define OFF_WHH0  196608u      // 1024*256 bf16 = 524288
#define OFF_WIH1  720896u      // 524288
#define OFF_WHH1  1245184u     // 524288
#define OFF_WFC   1769472u     // 80*256 bf16 = 40960
#define OFF_BIAS0 1810432u     // 1024 f32 = 4096
#define OFF_BIAS1 1814528u     // 4096
#define OFF_X0    1818624u     // 4096*96 bf16 = 786432
#define OFF_H0    2605056u     // 2*4096*256 bf16 = 4194304 (double buffered)
#define OFF_H1    6799360u     // 4194304 (double buffered, fallback mode)
#define OFF_C0    10993664u    // 4096*256 f32 = 4194304
#define OFF_C1    15187968u    // 4194304
#define OFF_H1ALL 19382272u    // optional: (MAXLEN+1)*4096*256 bf16 = 211812352
#define H1ALL_BYTES (211812352ull)
#define WS_NEED_BATCHED (OFF_H1ALL + H1ALL_BYTES)   // ~220.5 MB

// ---------------- device helpers ----------------
__device__ __forceinline__ v8f wmma_bf16(v16bf a, v16bf b, v8f c) {
    // D = A(16x32 bf16) * B(32x16 bf16) + C(16x16 f32)
    return __builtin_amdgcn_wmma_f32_16x16x32_bf16(
        /*neg_a=*/false, a, /*neg_b=*/false, b,
        /*c_mod=*/(short)0, c, /*reuse_a=*/false, /*reuse_b=*/false);
}

// A-fragment (16x32 bf16): lanes 0-15 hold K = base..base+7 and base+16..base+23
// (base = 0); lanes 16-31 the same rows with base = 8. p -> (row, k0 + base).
__device__ __forceinline__ v16bf load_afrag(const bf16_t* p) {
    v8bf lo = *(const v8bf*)p;
    v8bf hi = *(const v8bf*)(p + 16);
    v16bf a;
#pragma unroll
    for (int j = 0; j < 8; ++j) { a[j] = lo[j]; a[8 + j] = hi[j]; }
    return a;
}

// B-fragment (32x16 bf16): lane holds column n = lane&15, K = kb..kb+7 and
// kb+16..kb+23 with kb = (lane<16 ? 0 : 8). Two 16B loads (LDS- or global-friendly).
__device__ __forceinline__ v16bf load_bfrag(const bf16_t* p) {
    v8bf lo = *(const v8bf*)p;
    v8bf hi = *(const v8bf*)(p + 16);
    v16bf b;
#pragma unroll
    for (int j = 0; j < 8; ++j) { b[j] = lo[j]; b[8 + j] = hi[j]; }
    return b;
}

__device__ __forceinline__ float sigmoidf_(float x) {
    return 1.f / (1.f + __expf(-x));
}

// Async global -> LDS 16B transfer (ASYNCcnt tracked, no VGPR round trip).
__device__ __forceinline__ void async_b128(unsigned lds_off, const void* gsrc) {
    asm volatile("global_load_async_to_lds_b128 %0, %1, off"
                 :: "v"(lds_off), "v"(gsrc)
                 : "memory");
}
__device__ __forceinline__ void wait_async0() {
    asm volatile("s_wait_asynccnt 0x0" ::: "memory");
}

// ---------------- one-time prep kernels ----------------
__global__ void k_prep(const float* __restrict__ Wih0, const float* __restrict__ Whh0,
                       const float* __restrict__ Wih1, const float* __restrict__ Whh1,
                       const float* __restrict__ Wfc,
                       const float* __restrict__ bih0, const float* __restrict__ bhh0,
                       const float* __restrict__ bih1, const float* __restrict__ bhh1,
                       bf16_t* wih0b, bf16_t* whh0b, bf16_t* wih1b, bf16_t* whh1b,
                       bf16_t* wfcb, float* bias0, float* bias1) {
    int i = blockIdx.x * blockDim.x + threadIdx.x;
    int stride = gridDim.x * blockDim.x;
    for (int idx = i; idx < G4 * KIN; idx += stride) {       // zero-padded K
        int n = idx / KIN, k = idx % KIN;
        wih0b[idx] = (k < FEAT) ? (bf16_t)Wih0[n * FEAT + k] : (bf16_t)0.f;
    }
    for (int idx = i; idx < G4 * HID; idx += stride) {
        whh0b[idx] = (bf16_t)Whh0[idx];
        wih1b[idx] = (bf16_t)Wih1[idx];
        whh1b[idx] = (bf16_t)Whh1[idx];
    }
    for (int idx = i; idx < FEAT * HID; idx += stride) wfcb[idx] = (bf16_t)Wfc[idx];
    for (int idx = i; idx < G4; idx += stride) {
        bias0[idx] = bih0[idx] + bhh0[idx];
        bias1[idx] = bih1[idx] + bhh1[idx];
    }
}

__global__ void k_x0(const float* __restrict__ z, const float* __restrict__ Wli,
                     const float* __restrict__ bli, bf16_t* x0b) {
    int i = blockIdx.x * blockDim.x + threadIdx.x;
    int stride = gridDim.x * blockDim.x;
    for (int idx = i; idx < BATCH * KIN; idx += stride) {
        int b = idx / KIN, k = idx % KIN;
        float v = 0.f;
        if (k < FEAT) {
            v = bli[k];
            const float* zr = z + (size_t)b * LATENT;
            const float* wr = Wli + (size_t)k * LATENT;
#pragma unroll 4
            for (int j = 0; j < LATENT; ++j) v += zr[j] * wr[j];
        }
        x0b[idx] = (bf16_t)v;
    }
}

// zero h0 (both parities), c0, c1, and the initial h1 state (h1z, h1n elements)
__global__ void k_init(bf16_t* h0b, bf16_t* h1z, int h1n, float* c0, float* c1) {
    int i = blockIdx.x * blockDim.x + threadIdx.x;
    int stride = gridDim.x * blockDim.x;
    for (int idx = i; idx < 2 * BATCH * HID; idx += stride) h0b[idx] = (bf16_t)0.f;
    for (int idx = i; idx < h1n; idx += stride) h1z[idx] = (bf16_t)0.f;
    for (int idx = i; idx < BATCH * HID; idx += stride) {
        c0[idx] = 0.f;
        c1[idx] = 0.f;
    }
}

// ---------------- per-step kernels ----------------
// grid = (BATCH/128, HID/16), block = 256 threads (8 waves).
// Each wave owns a 16(batch) x 16(hidden) tile and computes ALL FOUR gate tiles
// with one shared A fragment. Weight rows for the block's j-tile are staged in
// LDS via async global->LDS copies. The K loop is software-pipelined by one
// 32-wide chunk: operands for chunk kc+1 are in flight while the four WMMAs of
// chunk kc execute, so s_wait_dscnt/loadcnt covers a full WMMA group of latency.
__global__ __launch_bounds__(256)
void k_step0(int t, const bf16_t* __restrict__ x0b, const float* __restrict__ tseq,
             const bf16_t* __restrict__ wih0b, const bf16_t* __restrict__ whh0b,
             const float* __restrict__ bias0,
             const bf16_t* __restrict__ h0_read, bf16_t* __restrict__ h0_write,
             float* __restrict__ c0) {
    __shared__ __align__(32) bf16_t wsm[4][16][K0TOT];   // 45056 B
    const int tid = threadIdx.x, lane = tid & 31, w = tid >> 5;

    // ---- stage W_ih0|W_hh0 rows for this j-tile (4 gates x 16 rows) ----
    for (int chunk = tid; chunk < 64 * 22; chunk += 256) {
        const int row = chunk / 22, part = chunk % 22;
        const int g = row >> 4, jn = row & 15;
        const int n = g * HID + blockIdx.y * 16 + jn;
        if (part < 6) {
            async_b128((unsigned)(size_t)&wsm[g][jn][part * 8],
                       wih0b + (size_t)n * KIN + part * 8);
        } else {
            const int p = part - 6;
            async_b128((unsigned)(size_t)&wsm[g][jn][KIN + p * 8],
                       whh0b + (size_t)n * HID + p * 8);
        }
    }
    wait_async0();
    __syncthreads();

    const int m0   = blockIdx.x * 128 + w * 16;
    const int mrow = m0 + (lane & 15);
    const int base = (lane < 16) ? 0 : 8;    // A-fragment K phase
    const int kb   = (lane < 16) ? 0 : 8;    // B-fragment K phase
    const int jn   = lane & 15;
    const bf16_t* hr = h0_read + (size_t)mrow * HID;
    const bf16_t* wp = &wsm[0][0][0];

    v8f acc[4] = {};
    v16bf acur, anxt;
    v16bf bcur[4], bnxt[4];

    // ---- phase 1: input projection, K = 0..95 ----
    if (t == 0) {
        const bf16_t* xr = x0b + (size_t)mrow * KIN;
        acur = load_afrag(xr + base);
#pragma unroll
        for (int g = 0; g < 4; ++g)
            bcur[g] = load_bfrag(wp + (g * 16 + jn) * K0TOT + kb);
#pragma unroll
        for (int kc = 0; kc < 3; ++kc) {
            if (kc < 2) {
                anxt = load_afrag(xr + (kc + 1) * 32 + base);
#pragma unroll
                for (int g = 0; g < 4; ++g)
                    bnxt[g] = load_bfrag(wp + (g * 16 + jn) * K0TOT + (kc + 1) * 32 + kb);
            } else {  // preload recurrent chunk 0
                anxt = load_afrag(hr + base);
#pragma unroll
                for (int g = 0; g < 4; ++g)
                    bnxt[g] = load_bfrag(wp + (g * 16 + jn) * K0TOT + KIN + kb);
            }
#pragma unroll
            for (int g = 0; g < 4; ++g) acc[g] = wmma_bf16(acur, bcur[g], acc[g]);
            acur = anxt;
#pragma unroll
            for (int g = 0; g < 4; ++g) bcur[g] = bnxt[g];
        }
    } else {
        const float* xr = tseq + ((size_t)mrow * MAXLEN + (t - 1)) * FEAT;
#pragma unroll
        for (int j = 0; j < 8; ++j) {
            int k1 = base + j, k2 = base + 16 + j;
            acur[j]     = (k1 < FEAT) ? (bf16_t)xr[k1] : (bf16_t)0.f;
            acur[8 + j] = (k2 < FEAT) ? (bf16_t)xr[k2] : (bf16_t)0.f;
        }
#pragma unroll
        for (int g = 0; g < 4; ++g)
            bcur[g] = load_bfrag(wp + (g * 16 + jn) * K0TOT + kb);
#pragma unroll
        for (int kc = 0; kc < 3; ++kc) {
            if (kc < 2) {
                const int k0 = (kc + 1) * 32;
#pragma unroll
                for (int j = 0; j < 8; ++j) {
                    int k1 = k0 + base + j, k2 = k0 + base + 16 + j;
                    anxt[j]     = (k1 < FEAT) ? (bf16_t)xr[k1] : (bf16_t)0.f;
                    anxt[8 + j] = (k2 < FEAT) ? (bf16_t)xr[k2] : (bf16_t)0.f;
                }
#pragma unroll
                for (int g = 0; g < 4; ++g)
                    bnxt[g] = load_bfrag(wp + (g * 16 + jn) * K0TOT + k0 + kb);
            } else {  // preload recurrent chunk 0
                anxt = load_afrag(hr + base);
#pragma unroll
                for (int g = 0; g < 4; ++g)
                    bnxt[g] = load_bfrag(wp + (g * 16 + jn) * K0TOT + KIN + kb);
            }
#pragma unroll
            for (int g = 0; g < 4; ++g) acc[g] = wmma_bf16(acur, bcur[g], acc[g]);
            acur = anxt;
#pragma unroll
            for (int g = 0; g < 4; ++g) bcur[g] = bnxt[g];
        }
    }

    // ---- phase 2: recurrent, K = 256 (chunk 0 already in acur/bcur) ----
#pragma unroll
    for (int kc = 0; kc < 8; ++kc) {
        if (kc < 7) {
            anxt = load_afrag(hr + (kc + 1) * 32 + base);
#pragma unroll
            for (int g = 0; g < 4; ++g)
                bnxt[g] = load_bfrag(wp + (g * 16 + jn) * K0TOT + KIN + (kc + 1) * 32 + kb);
        }
#pragma unroll
        for (int g = 0; g < 4; ++g) acc[g] = wmma_bf16(acur, bcur[g], acc[g]);
        if (kc < 7) {
            acur = anxt;
#pragma unroll
            for (int g = 0; g < 4; ++g) bcur[g] = bnxt[g];
        }
    }

    // ---- LSTM cell, fully in-register (lane: n = jj fixed, m = r + 8*(lane>=16))
    const int jj = blockIdx.y * 16 + jn;
    const float bi = bias0[jj];
    const float bf = bias0[HID + jj];
    const float bg = bias0[2 * HID + jj];
    const float bo = bias0[3 * HID + jj];
#pragma unroll
    for (int r = 0; r < 8; ++r) {
        const int b = m0 + r + 8 * (lane >> 4);
        float iv = sigmoidf_(acc[0][r] + bi);
        float fv = sigmoidf_(acc[1][r] + bf);
        float gv = tanhf(acc[2][r] + bg);
        float ov = sigmoidf_(acc[3][r] + bo);
        const size_t ci = (size_t)b * HID + jj;
        float cn = fv * c0[ci] + iv * gv;
        c0[ci] = cn;
        h0_write[ci] = (bf16_t)(ov * tanhf(cn));
    }
}

__global__ __launch_bounds__(256)
void k_step1(const bf16_t* __restrict__ h0_new, const bf16_t* __restrict__ h1_read,
             const bf16_t* __restrict__ wih1b, const bf16_t* __restrict__ whh1b,
             const float* __restrict__ bias1,
             bf16_t* __restrict__ h1_write, float* __restrict__ c1) {
    __shared__ __align__(32) bf16_t wsm[4][16][K1TOT];   // 65536 B
    const int tid = threadIdx.x, lane = tid & 31, w = tid >> 5;

    // ---- stage W_ih1|W_hh1 rows: per row 16 + 16 b128 chunks; 64 rows ----
    for (int chunk = tid; chunk < 64 * 32; chunk += 256) {
        const int row = chunk / 32, part = chunk % 32;
        const int g = row >> 4, jn = row & 15;
        const int n = g * HID + blockIdx.y * 16 + jn;
        if (part < 16) {
            async_b128((unsigned)(size_t)&wsm[g][jn][part * 8],
                       wih1b + (size_t)n * HID + part * 8);
        } else {
            const int p = part - 16;
            async_b128((unsigned)(size_t)&wsm[g][jn][HID + p * 8],
                       whh1b + (size_t)n * HID + p * 8);
        }
    }
    wait_async0();
    __syncthreads();

    const int m0   = blockIdx.x * 128 + w * 16;
    const int mrow = m0 + (lane & 15);
    const int base = (lane < 16) ? 0 : 8;
    const int kb   = (lane < 16) ? 0 : 8;
    const int jn   = lane & 15;
    const bf16_t* h0r = h0_new + (size_t)mrow * HID;
    const bf16_t* h1r = h1_read + (size_t)mrow * HID;
    const bf16_t* wp = &wsm[0][0][0];

    v8f acc[4] = {};
    v16bf acur, anxt;
    v16bf bcur[4], bnxt[4];

    acur = load_afrag(h0r + base);
#pragma unroll
    for (int g = 0; g < 4; ++g)
        bcur[g] = load_bfrag(wp + (g * 16 + jn) * K1TOT + kb);

    // ---- phase 1: W_ih1 * h0, K = 256 ----
#pragma unroll
    for (int kc = 0; kc < 8; ++kc) {
        if (kc < 7) {
            anxt = load_afrag(h0r + (kc + 1) * 32 + base);
#pragma unroll
            for (int g = 0; g < 4; ++g)
                bnxt[g] = load_bfrag(wp + (g * 16 + jn) * K1TOT + (kc + 1) * 32 + kb);
        } else {  // preload phase-2 chunk 0
            anxt = load_afrag(h1r + base);
#pragma unroll
            for (int g = 0; g < 4; ++g)
                bnxt[g] = load_bfrag(wp + (g * 16 + jn) * K1TOT + HID + kb);
        }
#pragma unroll
        for (int g = 0; g < 4; ++g) acc[g] = wmma_bf16(acur, bcur[g], acc[g]);
        acur = anxt;
#pragma unroll
        for (int g = 0; g < 4; ++g) bcur[g] = bnxt[g];
    }

    // ---- phase 2: W_hh1 * h1, K = 256 ----
#pragma unroll
    for (int kc = 0; kc < 8; ++kc) {
        if (kc < 7) {
            anxt = load_afrag(h1r + (kc + 1) * 32 + base);
#pragma unroll
            for (int g = 0; g < 4; ++g)
                bnxt[g] = load_bfrag(wp + (g * 16 + jn) * K1TOT + HID + (kc + 1) * 32 + kb);
        }
#pragma unroll
        for (int g = 0; g < 4; ++g) acc[g] = wmma_bf16(acur, bcur[g], acc[g]);
        if (kc < 7) {
            acur = anxt;
#pragma unroll
            for (int g = 0; g < 4; ++g) bcur[g] = bnxt[g];
        }
    }

    const int jj = blockIdx.y * 16 + jn;
    const float bi = bias1[jj];
    const float bf = bias1[HID + jj];
    const float bg = bias1[2 * HID + jj];
    const float bo = bias1[3 * HID + jj];
#pragma unroll
    for (int r = 0; r < 8; ++r) {
        const int b = m0 + r + 8 * (lane >> 4);
        float iv = sigmoidf_(acc[0][r] + bi);
        float fv = sigmoidf_(acc[1][r] + bf);
        float gv = tanhf(acc[2][r] + bg);
        float ov = sigmoidf_(acc[3][r] + bo);
        const size_t ci = (size_t)b * HID + jj;
        float cn = fv * c1[ci] + iv * gv;
        c1[ci] = cn;
        h1_write[ci] = (bf16_t)(ov * tanhf(cn));
    }
}

// Per-step FC fallback: y[b, t, :] = h1 @ W_fc^T + b_fc. 1280 tiles, 160 blocks.
__global__ __launch_bounds__(256)
void k_fc(int t, const bf16_t* __restrict__ h1b, const bf16_t* __restrict__ wfcb,
          const float* __restrict__ bfc, float* __restrict__ out) {
    const int tid = threadIdx.x, lane = tid & 31, w = tid >> 5;
    const int tile = blockIdx.x * 8 + w;
    const int mt = tile / 5, nt = tile % 5;
    const int mrow = mt * 16 + (lane & 15);
    const int base = (lane < 16) ? 0 : 8;
    const int kb   = (lane < 16) ? 0 : 8;
    const int ncol = nt * 16 + (lane & 15);

    const bf16_t* hr = h1b + (size_t)mrow * HID;
    const bf16_t* wr = wfcb + (size_t)ncol * HID;
    __builtin_prefetch(wr, 0, 0);   // global_prefetch_b8
    v8f acc = {};
    v16bf acur = load_afrag(hr + base);
    v16bf bcur = load_bfrag(wr + kb);
#pragma unroll
    for (int kc = 0; kc < 8; ++kc) {
        v16bf anxt, bnxt;
        if (kc < 7) {
            anxt = load_afrag(hr + (kc + 1) * 32 + base);
            bnxt = load_bfrag(wr + (kc + 1) * 32 + kb);
        }
        acc = wmma_bf16(acur, bcur, acc);
        if (kc < 7) { acur = anxt; bcur = bnxt; }
    }
#pragma unroll
    for (int r = 0; r < 8; ++r) {
        const int b = mt * 16 + r + 8 * (lane >> 4);
        const int f = nt * 16 + (lane & 15);
        out[((size_t)b * MAXLEN + t) * FEAT + f] = acc[r] + bfc[f];
    }
}

// Batched FC: one GEMM over all timesteps. h1all = [MAXLEN][BATCH][HID] bf16
// (slots 1..MAXLEN of the state buffer). M = MAXLEN*BATCH = 409600 rows.
// Tiles: 25600 m-tiles x 5 n-tiles = 128000; 8 per block -> 16000 blocks.
__global__ __launch_bounds__(256)
void k_fc_all(const bf16_t* __restrict__ h1all, const bf16_t* __restrict__ wfcb,
              const float* __restrict__ bfc, float* __restrict__ out) {
    const int tid = threadIdx.x, lane = tid & 31, w = tid >> 5;
    const int tile = blockIdx.x * 8 + w;
    const int mt = tile / 5, nt = tile % 5;
    const int mrow = mt * 16 + (lane & 15);      // row in [0, MAXLEN*BATCH)
    const int base = (lane < 16) ? 0 : 8;
    const int kb   = (lane < 16) ? 0 : 8;
    const int ncol = nt * 16 + (lane & 15);

    const bf16_t* hr = h1all + (size_t)mrow * HID;
    const bf16_t* wr = wfcb + (size_t)ncol * HID;
    __builtin_prefetch(wr, 0, 0);
    v8f acc = {};
    v16bf acur = load_afrag(hr + base);
    v16bf bcur = load_bfrag(wr + kb);
#pragma unroll
    for (int kc = 0; kc < 8; ++kc) {
        v16bf anxt, bnxt;
        if (kc < 7) {
            anxt = load_afrag(hr + (kc + 1) * 32 + base);
            bnxt = load_bfrag(wr + (kc + 1) * 32 + kb);
        }
        acc = wmma_bf16(acur, bcur, acc);
        if (kc < 7) { acur = anxt; bcur = bnxt; }
    }
#pragma unroll
    for (int r = 0; r < 8; ++r) {
        const int gr = mt * 16 + r + 8 * (lane >> 4);   // = t*BATCH + b
        const int t = gr >> 12, b = gr & (BATCH - 1);
        const int f = nt * 16 + (lane & 15);
        out[((size_t)b * MAXLEN + t) * FEAT + f] = acc[r] + bfc[f];
    }
}

// ---------------- launch ----------------
extern "C" void kernel_launch(void* const* d_in, const int* in_sizes, int n_in,
                              void* d_out, int out_size, void* d_ws, size_t ws_size,
                              hipStream_t stream) {
    const float* z    = (const float*)d_in[0];
    const float* tseq = (const float*)d_in[1];
    const float* Wli  = (const float*)d_in[2];
    const float* bli  = (const float*)d_in[3];
    const float* Wih0 = (const float*)d_in[4];
    const float* Whh0 = (const float*)d_in[5];
    const float* bih0 = (const float*)d_in[6];
    const float* bhh0 = (const float*)d_in[7];
    const float* Wih1 = (const float*)d_in[8];
    const float* Whh1 = (const float*)d_in[9];
    const float* bih1 = (const float*)d_in[10];
    const float* bhh1 = (const float*)d_in[11];
    const float* Wfc  = (const float*)d_in[12];
    const float* bfc  = (const float*)d_in[13];

    char* ws = (char*)d_ws;
    bf16_t* wih0b = (bf16_t*)(ws + OFF_WIH0);
    bf16_t* whh0b = (bf16_t*)(ws + OFF_WHH0);
    bf16_t* wih1b = (bf16_t*)(ws + OFF_WIH1);
    bf16_t* whh1b = (bf16_t*)(ws + OFF_WHH1);
    bf16_t* wfcb  = (bf16_t*)(ws + OFF_WFC);
    float*  bias0 = (float*)(ws + OFF_BIAS0);
    float*  bias1 = (float*)(ws + OFF_BIAS1);
    bf16_t* x0b   = (bf16_t*)(ws + OFF_X0);
    bf16_t* h0b   = (bf16_t*)(ws + OFF_H0);   // [2][BATCH][HID]
    bf16_t* h1b   = (bf16_t*)(ws + OFF_H1);   // [2][BATCH][HID] (fallback)
    float*  c0    = (float*)(ws + OFF_C0);
    float*  c1    = (float*)(ws + OFF_C1);
    bf16_t* h1all = (bf16_t*)(ws + OFF_H1ALL); // [MAXLEN+1][BATCH][HID] (optional)
    float*  out   = (float*)d_out;

    const bool batched = (ws_size >= WS_NEED_BATCHED);
    const size_t HS = (size_t)BATCH * HID;

    k_prep<<<512, 256, 0, stream>>>(Wih0, Whh0, Wih1, Whh1, Wfc,
                                    bih0, bhh0, bih1, bhh1,
                                    wih0b, whh0b, wih1b, whh1b, wfcb, bias0, bias1);
    k_x0<<<(BATCH * KIN + 255) / 256, 256, 0, stream>>>(z, Wli, bli, x0b);
    if (batched) {
        k_init<<<1024, 256, 0, stream>>>(h0b, h1all, (int)HS, c0, c1);
    } else {
        k_init<<<1024, 256, 0, stream>>>(h0b, h1b, (int)(2 * HS), c0, c1);
    }

    dim3 grid(BATCH / 128, HID / 16);
    for (int t = 0; t < MAXLEN; ++t) {
        const int rp = t & 1, wp = rp ^ 1;
        k_step0<<<grid, 256, 0, stream>>>(t, x0b, tseq, wih0b, whh0b, bias0,
                                          h0b + rp * HS, h0b + wp * HS, c0);
        if (batched) {
            // h1 state lives in h1all: read slot t, write slot t+1
            k_step1<<<grid, 256, 0, stream>>>(h0b + wp * HS, h1all + (size_t)t * HS,
                                              wih1b, whh1b, bias1,
                                              h1all + (size_t)(t + 1) * HS, c1);
        } else {
            k_step1<<<grid, 256, 0, stream>>>(h0b + wp * HS, h1b + rp * HS,
                                              wih1b, whh1b, bias1,
                                              h1b + wp * HS, c1);
            k_fc<<<160, 256, 0, stream>>>(t, h1b + wp * HS, wfcb, bfc, out);
        }
    }
    if (batched) {
        k_fc_all<<<16000, 256, 0, stream>>>(h1all + HS, wfcb, bfc, out);
    }
}
